// HSICLoss_25091198943287
// MI455X (gfx1250) — matrix operand order repaired and checked by reading, not compiled, not run on hardware
//
#include <hip/hip_runtime.h>

// ---------------------------------------------------------------------------
// HSIC(x, y), M=8192, D=128, sigma_x = sigma_y = 1.
// hsic = [ sum(K.*L) - (2/m) dot(sK,sL) + sumK*sumL/m^2 ] / (m-1)^2
//
// Strategy (MI455X / gfx1250):
//  * Fused: K/L never materialized (would be 2x256MB of HBM traffic).
//  * Pairwise dots via bf16 split-WMMA: x = hi + lo, S ~= hi*hi + hi*lo + lo*hi
//    -> fp32-grade accuracy at bf16 matrix-core rate (v_wmma_f32_16x16x32_bf16).
//  * bf16 hi/lo split precomputed ONCE into workspace (8MB, L2-resident):
//    removes ~2G redundant v_cvt ops from the hot loop.
//  * Symmetry: only lower-triangular tiles computed (2x less WMMA work);
//    off-diagonal tiles contribute row sums to rows i and column sums to
//    rows j, and 2x to sum(K.*L).
//  * Fallback path (on-the-fly conversion) if ws_size is too small.
// ---------------------------------------------------------------------------

typedef __attribute__((ext_vector_type(16))) __bf16 v16bf;
typedef __attribute__((ext_vector_type(8)))  __bf16 v8bf;
typedef __attribute__((ext_vector_type(4)))  __bf16 v4bf;
typedef __attribute__((ext_vector_type(8)))  float  v8f;

#define HSIC_M 8192
#define HSIC_D 128
#define LOG2E  1.4426950408889634f
#define ARR_ELEMS (HSIC_M * HSIC_D)          // 1,048,576

// Workspace float-index layout:
#define WS_ROWK   16
#define WS_ROWL   (16 + HSIC_M)
#define WS_NX     (16 + 2 * HSIC_M)
#define WS_NY     (16 + 3 * HSIC_M)
#define WS_ZERON  (16 + 2 * HSIC_M)          // klsum + both rowsum arrays
// bf16 region (byte offset, 256-aligned): xhi, xlo, yhi, ylo
#define BF_BASE_BYTES 131584
#define WS_NEED_BYTES (BF_BASE_BYTES + 4ull * ARR_ELEMS * 2ull)   // ~8.5 MB

// ============================ common kernels ===============================

__global__ void hsic_zero_kernel(float* __restrict__ ws) {
    int i = blockIdx.x * blockDim.x + threadIdx.x;
    if (i < WS_ZERON) ws[i] = 0.0f;
}

// One wave per row: squared norms of x rows and y rows.
__global__ __launch_bounds__(256) void hsic_norms_kernel(
    const float* __restrict__ x, const float* __restrict__ y,
    float* __restrict__ ws) {
    const int lane = threadIdx.x & 31;
    const int wave = threadIdx.x >> 5;
    const int row  = blockIdx.x * 8 + wave;
    const float* src = blockIdx.y ? y : x;
    float* dst       = ws + (blockIdx.y ? WS_NY : WS_NX);
    const float4* p = (const float4*)(src + (size_t)row * HSIC_D) + lane;
    float4 v = *p;
    float s = v.x * v.x + v.y * v.y + v.z * v.z + v.w * v.w;
#pragma unroll
    for (int m = 1; m < 32; m <<= 1) s += __shfl_xor(s, m, 32);
    if (lane == 0) dst[row] = s;
}

__global__ __launch_bounds__(256) void hsic_final_kernel(
    const float* __restrict__ ws, float* __restrict__ out) {
    __shared__ float sdot[256], ssk[256], ssl[256];
    const float* rowK = ws + WS_ROWK;
    const float* rowL = ws + WS_ROWL;
    float dot = 0.0f, sk = 0.0f, sl = 0.0f;
    for (int i = threadIdx.x; i < HSIC_M; i += 256) {
        float a = rowK[i], b = rowL[i];
        dot += a * b; sk += a; sl += b;
    }
    const int tid = threadIdx.x;
    sdot[tid] = dot; ssk[tid] = sk; ssl[tid] = sl;
    __syncthreads();
    for (int s = 128; s > 0; s >>= 1) {
        if (tid < s) {
            sdot[tid] += sdot[tid + s];
            ssk[tid]  += ssk[tid + s];
            ssl[tid]  += ssl[tid + s];
        }
        __syncthreads();
    }
    if (tid == 0) {
        const double m  = (double)HSIC_M;
        const double kl = (double)ws[0];
        double r = (kl - (2.0 / m) * (double)sdot[0]
                    + ((double)ssk[0] * (double)ssl[0]) / (m * m))
                   / ((m - 1.0) * (m - 1.0));
        out[0] = (float)r;
    }
}

// ===================== main path: precomputed bf16 split ===================

// Split fp32 x/y into bf16 hi/lo arrays once. blockIdx.y: 0 -> x, 1 -> y.
__global__ __launch_bounds__(256) void hsic_split_kernel(
    const float* __restrict__ x, const float* __restrict__ y,
    __bf16* __restrict__ bfbase) {
    const float* src = blockIdx.y ? y : x;
    __bf16* hi = bfbase + (size_t)blockIdx.y * (2 * ARR_ELEMS);
    __bf16* lo = hi + ARR_ELEMS;
    const int idx = (blockIdx.x * 256 + threadIdx.x) * 4;
    float4 v = *(const float4*)(src + idx);
    float f[4] = {v.x, v.y, v.z, v.w};
    v4bf hv, lv;
#pragma unroll
    for (int e = 0; e < 4; ++e) {
        __bf16 hb = (__bf16)f[e];
        hv[e] = hb;
        lv[e] = (__bf16)(f[e] - (float)hb);
    }
    *(v4bf*)(hi + idx) = hv;
    *(v4bf*)(lo + idx) = lv;
}

// A-fragment (contiguous k order kb..kb+15, not AMD's table order): two contiguous 8-elem chunks.
__device__ __forceinline__ v16bf load_a_frag(const __bf16* __restrict__ rowp,
                                             int kb) {
    v8bf a = *(const v8bf*)(rowp + kb);
    v8bf b = *(const v8bf*)(rowp + kb + 8);
    return __builtin_shufflevector(a, b, 0, 1, 2, 3, 4, 5, 6, 7,
                                         8, 9, 10, 11, 12, 13, 14, 15);
}

// Triangular fused tile kernel.
// grid = (256, 4), block = 256 (8 waves).
// Block bx processes strip pair {bx, 511-bx} (constant 513 tiles/pair).
// Partition p = by*8 + wave in [0,32); wave handles jt = p, p+32, ... <= it.
__global__ __launch_bounds__(256) void hsic_tiles_tri_kernel(
    const __bf16* __restrict__ bfbase, float* __restrict__ ws) {
    const int lane = threadIdx.x & 31;
    const int wave = threadIdx.x >> 5;
    const int h    = lane >> 4;               // half-wave (C layout: M += 8)
    const int ln   = lane & 15;
    const int p    = blockIdx.y * 8 + wave;   // 0..31

    const __bf16* xh = bfbase;
    const __bf16* xl = bfbase + ARR_ELEMS;
    const __bf16* yh = bfbase + 2 * ARR_ELEMS;
    const __bf16* yl = bfbase + 3 * ARR_ELEMS;

    float* klsum = ws;
    float* rowK  = ws + WS_ROWK;
    float* rowL  = ws + WS_ROWL;
    const float* nx = ws + WS_NX;
    const float* ny = ws + WS_NY;

    float klacc = 0.0f;

#pragma unroll 1
    for (int sidx = 0; sidx < 2; ++sidx) {
        const int it = sidx ? (511 - (int)blockIdx.x) : (int)blockIdx.x;
        const int i0 = it * 16;
        if (p > it) continue;                 // no tiles for this wave/strip

        // A-fragments for this strip: resident across the jt loop.
        v16bf axh[4], axl[4], ayh[4], ayl[4];
        const __bf16* xrh = xh + (size_t)(i0 + ln) * HSIC_D;
        const __bf16* xrl = xl + (size_t)(i0 + ln) * HSIC_D;
        const __bf16* yrh = yh + (size_t)(i0 + ln) * HSIC_D;
        const __bf16* yrl = yl + (size_t)(i0 + ln) * HSIC_D;
#pragma unroll
        for (int c = 0; c < 4; ++c) {
            const int kb = c * 32 + h * 16;
            axh[c] = load_a_frag(xrh, kb);
            axl[c] = load_a_frag(xrl, kb);
            ayh[c] = load_a_frag(yrh, kb);
            ayl[c] = load_a_frag(yrl, kb);
        }
        float nxi[8], nyi[8];
#pragma unroll
        for (int v = 0; v < 8; ++v) {
            nxi[v] = nx[i0 + v + 8 * h];
            nyi[v] = ny[i0 + v + 8 * h];
        }
        float rKacc[8], rLacc[8];
#pragma unroll
        for (int v = 0; v < 8; ++v) { rKacc[v] = 0.0f; rLacc[v] = 0.0f; }

#pragma unroll 1
        for (int jt = p; jt <= it; jt += 32) {
            const bool diag = (jt == it);
            const int j0 = jt * 16;
            const __bf16* bxh = xh + (size_t)(j0 + ln) * HSIC_D;
            const __bf16* bxl = xl + (size_t)(j0 + ln) * HSIC_D;
            const __bf16* byh = yh + (size_t)(j0 + ln) * HSIC_D;
            const __bf16* byl = yl + (size_t)(j0 + ln) * HSIC_D;

            v8f Sx = {};
            v8f Sy = {};
#pragma unroll
            for (int c = 0; c < 4; ++c) {
                const int kb = c * 32 + h * 16;   // B: 16 consecutive K / lane
                v16bf bh = *(const v16bf*)(bxh + kb);
                v16bf bl = *(const v16bf*)(bxl + kb);
                Sx = __builtin_amdgcn_wmma_f32_16x16x32_bf16(false, axh[c], false, bh, (short)0, Sx, false, false);
                Sx = __builtin_amdgcn_wmma_f32_16x16x32_bf16(false, axh[c], false, bl, (short)0, Sx, false, false);
                Sx = __builtin_amdgcn_wmma_f32_16x16x32_bf16(false, axl[c], false, bh, (short)0, Sx, false, false);
                bh = *(const v16bf*)(byh + kb);
                bl = *(const v16bf*)(byl + kb);
                Sy = __builtin_amdgcn_wmma_f32_16x16x32_bf16(false, ayh[c], false, bh, (short)0, Sy, false, false);
                Sy = __builtin_amdgcn_wmma_f32_16x16x32_bf16(false, ayh[c], false, bl, (short)0, Sy, false, false);
                Sy = __builtin_amdgcn_wmma_f32_16x16x32_bf16(false, ayl[c], false, bh, (short)0, Sy, false, false);
            }

            const float nxj = nx[j0 + ln];
            const float nyj = ny[j0 + ln];
            float tkl = 0.0f, colK = 0.0f, colL = 0.0f;
#pragma unroll
            for (int v = 0; v < 8; ++v) {
                // K = exp(-(d2)) = exp2(LOG2E * (2S - ni - nj))
                float Kv = exp2f(LOG2E * (2.0f * Sx[v] - nxi[v] - nxj));
                float Lv = exp2f(LOG2E * (2.0f * Sy[v] - nyi[v] - nyj));
                tkl      += Kv * Lv;
                rKacc[v] += Kv;
                rLacc[v] += Lv;
                colK     += Kv;
                colL     += Lv;
            }
            klacc += diag ? tkl : 2.0f * tkl;
            if (!diag) {
                // column sums -> rows j (symmetry). Combine the two halves
                // (lane l holds rows 0..7, lane l^16 holds rows 8..15).
                colK += __shfl_xor(colK, 16, 32);
                colL += __shfl_xor(colL, 16, 32);
                if (h == 0) {
                    atomicAdd(&rowK[j0 + ln], colK);
                    atomicAdd(&rowL[j0 + ln], colL);
                }
            }
        }

        // Row-sum partials -> rows i: reduce across 16 lanes of each half.
#pragma unroll
        for (int v = 0; v < 8; ++v) {
            float rk = rKacc[v], rl = rLacc[v];
#pragma unroll
            for (int m = 1; m < 16; m <<= 1) {
                rk += __shfl_xor(rk, m, 32);
                rl += __shfl_xor(rl, m, 32);
            }
            if (ln == 0) {                    // lanes 0 and 16
                atomicAdd(&rowK[i0 + v + 8 * h], rk);
                atomicAdd(&rowL[i0 + v + 8 * h], rl);
            }
        }
    }

#pragma unroll
    for (int m = 1; m < 32; m <<= 1) klacc += __shfl_xor(klacc, m, 32);
    if (lane == 0) atomicAdd(klsum, klacc);
}

// ============== fallback path: on-the-fly conversion (small ws) ============

__device__ __forceinline__ void split16(const float* __restrict__ p0,
                                        const float* __restrict__ p1,
                                        v16bf& hi, v16bf& lo) {
    const float4* q0 = (const float4*)p0;
    const float4* q1 = (const float4*)p1;
    float4 a = q0[0], b = q0[1], c = q1[0], d = q1[1];
    float f[16] = {a.x,a.y,a.z,a.w, b.x,b.y,b.z,b.w,
                   c.x,c.y,c.z,c.w, d.x,d.y,d.z,d.w};
#pragma unroll
    for (int e = 0; e < 16; ++e) {
        __bf16 hv = (__bf16)f[e];
        hi[e] = hv;
        lo[e] = (__bf16)(f[e] - (float)hv);
    }
}

__global__ __launch_bounds__(256) void hsic_tiles_fused_kernel(
    const float* __restrict__ x, const float* __restrict__ y,
    float* __restrict__ ws) {
    const int lane = threadIdx.x & 31;
    const int wave = threadIdx.x >> 5;
    const int h    = lane >> 4;
    const int ln   = lane & 15;
    const int i0   = blockIdx.x * 16;
    const int part = blockIdx.y * 8 + wave;

    float* klsum = ws;
    float* rowK  = ws + WS_ROWK;
    float* rowL  = ws + WS_ROWL;
    const float* nx = ws + WS_NX;
    const float* ny = ws + WS_NY;

    v16bf axh[4], axl[4], ayh[4], ayl[4];
    const float* xrow = x + (size_t)(i0 + ln) * HSIC_D;
    const float* yrow = y + (size_t)(i0 + ln) * HSIC_D;
#pragma unroll
    for (int c = 0; c < 4; ++c) {
        const int kb = c * 32 + h * 16;
        split16(xrow + kb, xrow + kb + 8, axh[c], axl[c]);
        split16(yrow + kb, yrow + kb + 8, ayh[c], ayl[c]);
    }
    float nxi[8], nyi[8];
#pragma unroll
    for (int v = 0; v < 8; ++v) {
        nxi[v] = nx[i0 + v + 8 * h];
        nyi[v] = ny[i0 + v + 8 * h];
    }
    float klacc = 0.0f;
    float rKacc[8], rLacc[8];
#pragma unroll
    for (int v = 0; v < 8; ++v) { rKacc[v] = 0.0f; rLacc[v] = 0.0f; }

    for (int t = 0; t < 8; ++t) {
        const int j0 = (part * 8 + t) * 16;
        const float* xjr = x + (size_t)(j0 + ln) * HSIC_D;
        const float* yjr = y + (size_t)(j0 + ln) * HSIC_D;
        v8f Sx = {};
        v8f Sy = {};
#pragma unroll
        for (int c = 0; c < 4; ++c) {
            const int kb = c * 32 + h * 16;
            v16bf bh, bl;
            split16(xjr + kb, xjr + kb + 8, bh, bl);
            Sx = __builtin_amdgcn_wmma_f32_16x16x32_bf16(false, axh[c], false, bh, (short)0, Sx, false, false);
            Sx = __builtin_amdgcn_wmma_f32_16x16x32_bf16(false, axh[c], false, bl, (short)0, Sx, false, false);
            Sx = __builtin_amdgcn_wmma_f32_16x16x32_bf16(false, axl[c], false, bh, (short)0, Sx, false, false);
            split16(yjr + kb, yjr + kb + 8, bh, bl);
            Sy = __builtin_amdgcn_wmma_f32_16x16x32_bf16(false, ayh[c], false, bh, (short)0, Sy, false, false);
            Sy = __builtin_amdgcn_wmma_f32_16x16x32_bf16(false, ayh[c], false, bl, (short)0, Sy, false, false);
            Sy = __builtin_amdgcn_wmma_f32_16x16x32_bf16(false, ayl[c], false, bh, (short)0, Sy, false, false);
        }
        const float nxj = nx[j0 + ln];
        const float nyj = ny[j0 + ln];
#pragma unroll
        for (int v = 0; v < 8; ++v) {
            float Kv = exp2f(LOG2E * (2.0f * Sx[v] - nxi[v] - nxj));
            float Lv = exp2f(LOG2E * (2.0f * Sy[v] - nyi[v] - nyj));
            klacc    += Kv * Lv;
            rKacc[v] += Kv;
            rLacc[v] += Lv;
        }
    }
#pragma unroll
    for (int v = 0; v < 8; ++v) {
        float rk = rKacc[v], rl = rLacc[v];
#pragma unroll
        for (int m = 1; m < 16; m <<= 1) {
            rk += __shfl_xor(rk, m, 32);
            rl += __shfl_xor(rl, m, 32);
        }
        if (ln == 0) {
            atomicAdd(&rowK[i0 + v + 8 * h], rk);
            atomicAdd(&rowL[i0 + v + 8 * h], rl);
        }
    }
#pragma unroll
    for (int m = 1; m < 32; m <<= 1) klacc += __shfl_xor(klacc, m, 32);
    if (lane == 0) atomicAdd(klsum, klacc);
}

// ================================ launch ===================================

extern "C" void kernel_launch(void* const* d_in, const int* in_sizes, int n_in,
                              void* d_out, int out_size, void* d_ws, size_t ws_size,
                              hipStream_t stream) {
    const float* x = (const float*)d_in[0];
    const float* y = (const float*)d_in[1];
    float* ws  = (float*)d_ws;
    float* out = (float*)d_out;
    (void)in_sizes; (void)n_in; (void)out_size;

    hsic_zero_kernel<<<(WS_ZERON + 255) / 256, 256, 0, stream>>>(ws);
    hsic_norms_kernel<<<dim3(HSIC_M / 8, 2), 256, 0, stream>>>(x, y, ws);

    if (ws_size >= WS_NEED_BYTES) {
        __bf16* bfbase = (__bf16*)((char*)d_ws + BF_BASE_BYTES);
        hsic_split_kernel<<<dim3(ARR_ELEMS / 1024, 2), 256, 0, stream>>>(x, y, bfbase);
        hsic_tiles_tri_kernel<<<dim3(256, 4), 256, 0, stream>>>(bfbase, ws);
    } else {
        hsic_tiles_fused_kernel<<<dim3(HSIC_M / 16, 8), 256, 0, stream>>>(x, y, ws);
    }
    hsic_final_kernel<<<1, 256, 0, stream>>>(ws, out);
}
